// TinyViTBlock_15135464751771
// MI455X (gfx1250) — matrix operand order repaired and checked by reading, not compiled, not run on hardware
//
#include <hip/hip_runtime.h>
#include <hip/hip_bf16.h>

typedef __attribute__((ext_vector_type(16))) _Float16 v16h;
typedef __attribute__((ext_vector_type(8)))  _Float16 v8h;
typedef __attribute__((ext_vector_type(8)))  float    v8f;
typedef __attribute__((ext_vector_type(4)))  int      v4i;

#define DIMC   576
#define HEADS  18
#define HD     32
#define NW     49            // tokens per 7x7 window
#define MROWS  25088         // 32 * 784
#define HIDDEN 2304

#if __has_builtin(__builtin_amdgcn_global_load_async_to_lds_b128)
#define HAVE_ASYNC_LDS 1
typedef __attribute__((address_space(1))) v4i g_v4i;   // global int4
typedef __attribute__((address_space(3))) v4i l_v4i;   // LDS int4
#else
#define HAVE_ASYNC_LDS 0
#endif

__device__ __forceinline__ void wait_async_zero() {
#if HAVE_ASYNC_LDS
#if __has_builtin(__builtin_amdgcn_s_wait_asynccnt)
    __builtin_amdgcn_s_wait_asynccnt(0);
#else
    asm volatile("s_wait_asynccnt 0" ::: "memory");
#endif
#endif
}

__device__ __forceinline__ int iabs(int v) { return v < 0 ? -v : v; }

__device__ __forceinline__ v16h make_v16(const _Float16* p0, const _Float16* p1) {
    v8h lo = *(const v8h*)p0;
    v8h hi = *(const v8h*)p1;
    v16h r;
#pragma unroll
    for (int i = 0; i < 8; ++i) { r[i] = lo[i]; r[i + 8] = hi[i]; }
    return r;
}

// ---------------------------------------------------------------------------
// Weight convert + transpose: wt[n*K + k] = (f16) w[k*N + n]
// ---------------------------------------------------------------------------
__global__ __launch_bounds__(256) void wtr_kernel(const float* __restrict__ w,
                                                  _Float16* __restrict__ wt,
                                                  int K, int N) {
    for (size_t i = (size_t)blockIdx.x * 256 + threadIdx.x; i < (size_t)K * N;
         i += (size_t)gridDim.x * 256) {
        int n = (int)(i % N);
        size_t k = i / N;
        wt[(size_t)n * K + k] = (_Float16)w[i];
    }
}

// ---------------------------------------------------------------------------
// LayerNorm over C=576. One wave per token (8 waves / block).
// PERM=1: write rows in window-major order (Swin window gather).
// ---------------------------------------------------------------------------
template <int PERM>
__global__ __launch_bounds__(256) void ln_kernel(const float* __restrict__ x,
                                                 const float* __restrict__ g,
                                                 const float* __restrict__ b,
                                                 _Float16* __restrict__ out) {
    const int tok = blockIdx.x * 8 + (threadIdx.x >> 5);
    const int lane = threadIdx.x & 31;
    const float* xp = x + (size_t)tok * DIMC;

    float v[18];
    float s = 0.f;
#pragma unroll
    for (int i = 0; i < 18; ++i) { v[i] = xp[lane + 32 * i]; s += v[i]; }
#pragma unroll
    for (int m = 1; m < 32; m <<= 1) s += __shfl_xor(s, m, 32);
    const float mean = s * (1.0f / 576.0f);

    float q = 0.f;
#pragma unroll
    for (int i = 0; i < 18; ++i) { float d = v[i] - mean; q += d * d; }
#pragma unroll
    for (int m = 1; m < 32; m <<= 1) q += __shfl_xor(q, m, 32);
    const float rstd = rsqrtf(q * (1.0f / 576.0f) + 1e-5f);

    size_t row;
    if (PERM) {
        int bb = tok / 784, l = tok % 784;
        int hh = l / 28, ww = l % 28;
        row = ((size_t)bb * 16 + (hh / 7) * 4 + (ww / 7)) * NW + (hh % 7) * 7 + (ww % 7);
    } else {
        row = (size_t)tok;
    }
    _Float16* op = out + row * DIMC;
#pragma unroll
    for (int i = 0; i < 18; ++i) {
        int c = lane + 32 * i;
        op[c] = (_Float16)((v[i] - mean) * rstd * g[c] + b[c]);
    }
}

// ---------------------------------------------------------------------------
// WMMA GEMM: C[M,N] = A[M,K] (f16, row-major) * Bt[N,K]^T (f16) + bias
// Block tile 256x64, 8 waves; each wave owns 32x64 = 2x4 WMMA 16x16x32 tiles.
// Tiles staged global->LDS with GLOBAL_LOAD_ASYNC_TO_LDS_B128 when available
// (ASYNCcnt + s_wait_asynccnt), else via VGPR b128 copies.
// MODE 0: out f16 = acc+bias                       (qkv)
// MODE 1: out f16 = gelu(acc+bias)                 (fc1)
// MODE 2: out f32[invperm(row)] = acc+bias+resid   (proj, window un-gather)
// MODE 3: out f32 = acc+bias+resid                 (fc2)
// ---------------------------------------------------------------------------
template <int MODE>
__global__ __launch_bounds__(256) void gemm_wmma_k(
    const _Float16* __restrict__ A, const _Float16* __restrict__ Bt,
    const float* __restrict__ bias, _Float16* __restrict__ outH,
    float* __restrict__ outF, const float* __restrict__ resid,
    int M, int N, int K) {
    __shared__ __align__(16) _Float16 As[256][32];   // 16 KB
    __shared__ __align__(16) _Float16 Bs[64][32];    //  4 KB

    const int tid = threadIdx.x;
    const int lane = tid & 31;
    const int wv = tid >> 5;          // 0..7 : row band 32*wv
    const int lo = lane & 15, hi = lane >> 4;
    const int m0 = blockIdx.y * 256;
    const int n0 = blockIdx.x * 64;

    const int arow = tid >> 1, acol = (tid & 1) * 16;
    const int brow = tid >> 2, bcol = (tid & 3) * 8;
    const _Float16* Ap0 = A + (size_t)(m0 + arow) * K + acol;
    const _Float16* Ap1 = A + (size_t)(m0 + arow + 128) * K + acol;
    const _Float16* Bp  = Bt + (size_t)(n0 + brow) * K + bcol;

    v8f acc[2][4] = {};

    for (int kt = 0; kt < K; kt += 32) {
#if HAVE_ASYNC_LDS
        __builtin_amdgcn_global_load_async_to_lds_b128(
            (g_v4i*)(void*)(Ap0 + kt),     (l_v4i*)(void*)&As[arow][acol], 0, 0);
        __builtin_amdgcn_global_load_async_to_lds_b128(
            (g_v4i*)(void*)(Ap0 + kt + 8), (l_v4i*)(void*)&As[arow][acol + 8], 0, 0);
        __builtin_amdgcn_global_load_async_to_lds_b128(
            (g_v4i*)(void*)(Ap1 + kt),     (l_v4i*)(void*)&As[arow + 128][acol], 0, 0);
        __builtin_amdgcn_global_load_async_to_lds_b128(
            (g_v4i*)(void*)(Ap1 + kt + 8), (l_v4i*)(void*)&As[arow + 128][acol + 8], 0, 0);
        __builtin_amdgcn_global_load_async_to_lds_b128(
            (g_v4i*)(void*)(Bp + kt),      (l_v4i*)(void*)&Bs[brow][bcol], 0, 0);
        wait_async_zero();
#else
        *(v8h*)&As[arow][acol]           = *(const v8h*)(Ap0 + kt);
        *(v8h*)&As[arow][acol + 8]       = *(const v8h*)(Ap0 + kt + 8);
        *(v8h*)&As[arow + 128][acol]     = *(const v8h*)(Ap1 + kt);
        *(v8h*)&As[arow + 128][acol + 8] = *(const v8h*)(Ap1 + kt + 8);
        *(v8h*)&Bs[brow][bcol]           = *(const v8h*)(Bp + kt);
#endif
        if (kt + 32 < K) {
            __builtin_prefetch(Ap0 + kt + 32, 0, 1);   // global_prefetch_b8
            __builtin_prefetch(Ap1 + kt + 32, 0, 1);
            __builtin_prefetch(Bp + kt + 32, 0, 1);
        }
        __syncthreads();

        v16h af[2], bf[4];
#pragma unroll
        for (int ii = 0; ii < 2; ++ii) {
            int r = 32 * wv + 16 * ii + lo;
            af[ii] = make_v16(&As[r][8 * hi], &As[r][16 + 8 * hi]);
        }
#pragma unroll
        for (int jj = 0; jj < 4; ++jj) {
            int c = 16 * jj + lo;
            bf[jj] = make_v16(&Bs[c][8 * hi], &Bs[c][16 + 8 * hi]);
        }
#pragma unroll
        for (int ii = 0; ii < 2; ++ii)
#pragma unroll
            for (int jj = 0; jj < 4; ++jj)
                acc[ii][jj] = __builtin_amdgcn_wmma_f32_16x16x32_f16(
                    false, af[ii], false, bf[jj], (short)0, acc[ii][jj], false, false);
        __syncthreads();
    }

#pragma unroll
    for (int ii = 0; ii < 2; ++ii)
#pragma unroll
        for (int jj = 0; jj < 4; ++jj)
#pragma unroll
            for (int r = 0; r < 8; ++r) {
                int row = m0 + 32 * wv + 16 * ii + r + 8 * hi;
                int col = n0 + 16 * jj + lo;
                float v = acc[ii][jj][r] + bias[col];
                if (MODE == 0) {
                    outH[(size_t)row * N + col] = (_Float16)v;
                } else if (MODE == 1) {
                    v = 0.5f * v * (1.0f + erff(v * 0.70710678f));
                    outH[(size_t)row * N + col] = (_Float16)v;
                } else if (MODE == 2) {
                    int win = row / NW, tt = row % NW;
                    int b = win >> 4, wl = win & 15;
                    int hh = (wl >> 2) * 7 + tt / 7;
                    int ww = (wl & 3) * 7 + tt % 7;
                    size_t idx = ((size_t)b * 784 + hh * 28 + ww) * DIMC + col;
                    outF[idx] = v + resid[idx];
                } else {
                    size_t idx = (size_t)row * N + col;
                    outF[idx] = v + resid[idx];
                }
            }
}

// ---------------------------------------------------------------------------
// Windowed attention: one block per (window, head). 4 waves.
// q(49x32) k(49x32) v(49x32) padded to 64; scores + LDS-staged bias +
// softmax + a@v, all matmuls through v_wmma_f32_16x16x32_f16.
// ---------------------------------------------------------------------------
__global__ __launch_bounds__(128) void attn_kernel(const _Float16* __restrict__ qkv,
                                                   const float* __restrict__ biases,
                                                   _Float16* __restrict__ obuf) {
    __shared__ __align__(16) _Float16 q_s[64][32];
    __shared__ __align__(16) _Float16 k_s[64][32];
    __shared__ __align__(16) _Float16 vT_s[32][64];
    __shared__ __align__(16) _Float16 a_s[64][64];
    __shared__ float bias_s[NW];

    const int win = blockIdx.x / HEADS;
    const int head = blockIdx.x % HEADS;
    const int tid = threadIdx.x;
    const int lane = tid & 31;
    const int wv = tid >> 5;
    const int lo = lane & 15, hi = lane >> 4;

    if (tid < NW) bias_s[tid] = biases[head * NW + tid];

    if (tid < 64) {
        int tok = tid;
        if (tok < NW) {
            const _Float16* p = qkv + ((size_t)(win * NW + tok)) * (3 * DIMC) + head * 96;
            *(v8h*)&q_s[tok][0]  = *(const v8h*)(p + 0);
            *(v8h*)&q_s[tok][8]  = *(const v8h*)(p + 8);
            *(v8h*)&q_s[tok][16] = *(const v8h*)(p + 16);
            *(v8h*)&q_s[tok][24] = *(const v8h*)(p + 24);
            *(v8h*)&k_s[tok][0]  = *(const v8h*)(p + 32);
            *(v8h*)&k_s[tok][8]  = *(const v8h*)(p + 40);
            *(v8h*)&k_s[tok][16] = *(const v8h*)(p + 48);
            *(v8h*)&k_s[tok][24] = *(const v8h*)(p + 56);
            _Float16 vv[32];
            *(v8h*)&vv[0]  = *(const v8h*)(p + 64);
            *(v8h*)&vv[8]  = *(const v8h*)(p + 72);
            *(v8h*)&vv[16] = *(const v8h*)(p + 80);
            *(v8h*)&vv[24] = *(const v8h*)(p + 88);
#pragma unroll
            for (int d = 0; d < 32; ++d) vT_s[d][tok] = vv[d];
        } else {
            v8h z = {};
#pragma unroll
            for (int c = 0; c < 32; c += 8) {
                *(v8h*)&q_s[tok][c] = z;
                *(v8h*)&k_s[tok][c] = z;
            }
#pragma unroll
            for (int d = 0; d < 32; ++d) vT_s[d][tok] = (_Float16)0.f;
        }
    }
    __syncthreads();

    // scores = q @ k^T  (rows 16*wv .. 16*wv+15)
    v16h qa = make_v16(&q_s[16 * wv + lo][8 * hi], &q_s[16 * wv + lo][16 + 8 * hi]);
    v8f acc[4];
    v8f zc = {};
#pragma unroll
    for (int j = 0; j < 4; ++j) {
        v16h kb = make_v16(&k_s[16 * j + lo][8 * hi], &k_s[16 * j + lo][16 + 8 * hi]);
        acc[j] = __builtin_amdgcn_wmma_f32_16x16x32_f16(false, qa, false, kb,
                                                        (short)0, zc, false, false);
    }

    // Per-lane key-token coordinates (branch-free; clamp padded to 0)
    int rm[4], cm[4];
    bool mvalid[4];
#pragma unroll
    for (int j = 0; j < 4; ++j) {
        int m_tok = 16 * j + lo;
        mvalid[j] = m_tok < NW;
        int mt = mvalid[j] ? m_tok : 0;
        rm[j] = mt / 7;
        cm[j] = mt % 7;
    }

    // scale + relative-position bias + softmax over the 49 (padded 64) keys
    const float scale = 0.17677669529663687f;  // 1/sqrt(32)
    float sc[4][8];
    const int nbase = 16 * wv + 8 * hi;
#pragma unroll
    for (int r = 0; r < 8; ++r) {
        const int n_tok = nbase + r;
        const bool nvalid = n_tok < NW;
        const int nt = nvalid ? n_tok : 0;
        const int rn = nt / 7, cn = nt % 7;
        float rmax = -3e38f;
#pragma unroll
        for (int j = 0; j < 4; ++j) {
            float bv = bias_s[iabs(rn - rm[j]) * 7 + iabs(cn - cm[j])];
            float s = (nvalid && mvalid[j]) ? (acc[j][r] * scale + bv) : -1e30f;
            sc[j][r] = s;
            rmax = fmaxf(rmax, s);
        }
#pragma unroll
        for (int m = 1; m < 16; m <<= 1) rmax = fmaxf(rmax, __shfl_xor(rmax, m, 32));
        float rsum = 0.f;
#pragma unroll
        for (int j = 0; j < 4; ++j) {
            float e = __expf(sc[j][r] - rmax);
            sc[j][r] = e;
            rsum += e;
        }
#pragma unroll
        for (int m = 1; m < 16; m <<= 1) rsum += __shfl_xor(rsum, m, 32);
        const float inv = 1.0f / rsum;
#pragma unroll
        for (int j = 0; j < 4; ++j)
            a_s[n_tok][16 * j + lo] = (_Float16)(sc[j][r] * inv);
    }
    __syncthreads();

    // out = a @ v   (K = 64 keys, two 32-wide WMMA steps; N = 32 head dims)
#pragma unroll
    for (int j2 = 0; j2 < 2; ++j2) {
        v8f oc = {};
#pragma unroll
        for (int kb = 0; kb < 64; kb += 32) {
            v16h af = make_v16(&a_s[16 * wv + lo][kb + 8 * hi],
                               &a_s[16 * wv + lo][kb + 16 + 8 * hi]);
            v16h bf = make_v16(&vT_s[16 * j2 + lo][kb + 8 * hi],
                               &vT_s[16 * j2 + lo][kb + 16 + 8 * hi]);
            oc = __builtin_amdgcn_wmma_f32_16x16x32_f16(false, af, false, bf,
                                                        (short)0, oc, false, false);
        }
#pragma unroll
        for (int r = 0; r < 8; ++r) {
            int tok = 16 * wv + r + 8 * hi;
            if (tok < NW)
                obuf[((size_t)(win * NW + tok)) * DIMC + head * HD + 16 * j2 + lo] =
                    (_Float16)oc[r];
        }
    }
}

// ---------------------------------------------------------------------------
// Depthwise 3x3 conv + BatchNorm (inference). One thread per (b,l,c).
// ---------------------------------------------------------------------------
__global__ __launch_bounds__(256) void conv_bn_kernel(
    const float* __restrict__ x1, const float* __restrict__ cw,
    const float* __restrict__ bg, const float* __restrict__ bb,
    const float* __restrict__ bm, const float* __restrict__ bv,
    float* __restrict__ x2) {
    const size_t idx = (size_t)blockIdx.x * 256 + threadIdx.x;
    const int c = (int)(idx % DIMC);
    const size_t t = idx / DIMC;
    const int l = (int)(t % 784);
    const int bI = (int)(t / 784);
    const int hh = l / 28, ww = l % 28;
    float sum = 0.f;
#pragma unroll
    for (int dy = -1; dy <= 1; ++dy)
#pragma unroll
        for (int dx = -1; dx <= 1; ++dx) {
            int y = hh + dy, xq = ww + dx;
            if (y >= 0 && y < 28 && xq >= 0 && xq < 28)
                sum += cw[c * 9 + (dy + 1) * 3 + (dx + 1)] *
                       x1[((size_t)bI * 784 + y * 28 + xq) * DIMC + c];
        }
    const float inv = rsqrtf(bv[c] + 1e-5f);
    x2[idx] = (sum - bm[c]) * inv * bg[c] + bb[c];
}

// ---------------------------------------------------------------------------
extern "C" void kernel_launch(void* const* d_in, const int* in_sizes, int n_in,
                              void* d_out, int out_size, void* d_ws, size_t ws_size,
                              hipStream_t stream) {
    const float* x        = (const float*)d_in[0];
    const float* ln1_g    = (const float*)d_in[1];
    const float* ln1_b    = (const float*)d_in[2];
    const float* qkv_w    = (const float*)d_in[3];
    const float* qkv_b    = (const float*)d_in[4];
    const float* attn_bia = (const float*)d_in[5];
    const float* proj_w   = (const float*)d_in[6];
    const float* proj_b   = (const float*)d_in[7];
    const float* conv_w   = (const float*)d_in[8];
    const float* bn_g     = (const float*)d_in[9];
    const float* bn_b     = (const float*)d_in[10];
    const float* bn_mean  = (const float*)d_in[11];
    const float* bn_var   = (const float*)d_in[12];
    const float* ln2_g    = (const float*)d_in[13];
    const float* ln2_b    = (const float*)d_in[14];
    const float* fc1_w    = (const float*)d_in[15];
    const float* fc1_b    = (const float*)d_in[16];
    const float* fc2_w    = (const float*)d_in[17];
    const float* fc2_b    = (const float*)d_in[18];
    float* out            = (float*)d_out;

    char* ws = (char*)d_ws;
    size_t off = 0;
    auto take = [&](size_t bytes) -> char* {
        char* p = ws + off;
        off += (bytes + 255) & ~(size_t)255;
        return p;
    };
    const size_t M = MROWS;
    _Float16* wT_qkv = (_Float16*)take((size_t)3 * DIMC * DIMC * 2);
    _Float16* wT_prj = (_Float16*)take((size_t)DIMC * DIMC * 2);
    _Float16* wT_fc1 = (_Float16*)take((size_t)HIDDEN * DIMC * 2);
    _Float16* wT_fc2 = (_Float16*)take((size_t)DIMC * HIDDEN * 2);
    _Float16* hbuf   = (_Float16*)take(M * DIMC * 2);          // LN1 out, reused for LN2 out
    _Float16* bigbuf = (_Float16*)take(M * HIDDEN * 2);        // qkv out, reused for fc1 out
    _Float16* obuf   = (_Float16*)take(M * DIMC * 2);          // attention out
    float*    x1     = (float*)take(M * DIMC * 4);             // after proj + residual
    float*    x2     = (float*)take(M * DIMC * 4);             // after conv + BN
    (void)ws_size; (void)in_sizes; (void)n_in; (void)out_size;

    // Weight transpose + f16 convert
    wtr_kernel<<<2048, 256, 0, stream>>>(qkv_w, wT_qkv, DIMC, 3 * DIMC);
    wtr_kernel<<<2048, 256, 0, stream>>>(proj_w, wT_prj, DIMC, DIMC);
    wtr_kernel<<<2048, 256, 0, stream>>>(fc1_w, wT_fc1, DIMC, HIDDEN);
    wtr_kernel<<<2048, 256, 0, stream>>>(fc2_w, wT_fc2, HIDDEN, DIMC);

    // LN1 (window-gathered rows)
    ln_kernel<1><<<MROWS / 8, 256, 0, stream>>>(x, ln1_g, ln1_b, hbuf);
    // QKV GEMM: [25088,576] x [576,1728]
    gemm_wmma_k<0><<<dim3(27, 98), 256, 0, stream>>>(hbuf, wT_qkv, qkv_b, bigbuf,
                                                     nullptr, nullptr, (int)M, 3 * DIMC, DIMC);
    // Windowed attention
    attn_kernel<<<512 * HEADS, 128, 0, stream>>>(bigbuf, attn_bia, obuf);
    // Proj GEMM + residual (window un-gather)
    gemm_wmma_k<2><<<dim3(9, 98), 256, 0, stream>>>(obuf, wT_prj, proj_b, nullptr,
                                                    x1, x, (int)M, DIMC, DIMC);
    // Depthwise conv + BN
    conv_bn_kernel<<<(unsigned)(M * DIMC / 256), 256, 0, stream>>>(x1, conv_w, bn_g, bn_b,
                                                                   bn_mean, bn_var, x2);
    // LN2
    ln_kernel<0><<<MROWS / 8, 256, 0, stream>>>(x2, ln2_g, ln2_b, hbuf);
    // MLP
    gemm_wmma_k<1><<<dim3(36, 98), 256, 0, stream>>>(hbuf, wT_fc1, fc1_b, bigbuf,
                                                     nullptr, nullptr, (int)M, HIDDEN, DIMC);
    gemm_wmma_k<3><<<dim3(9, 98), 256, 0, stream>>>(bigbuf, wT_fc2, fc2_b, nullptr,
                                                    out, x2, (int)M, DIMC, HIDDEN);
}
